// SortingLayer_29506425323656
// MI455X (gfx1250) — matrix operand order repaired and checked by reading, not compiled, not run on hardware
//
#include <hip/hip_runtime.h>
#include <hip/hip_bf16.h>

typedef __attribute__((ext_vector_type(16))) _Float16 v16h;
typedef __attribute__((ext_vector_type(8)))  _Float16 v8h;
typedef __attribute__((ext_vector_type(8)))  float    v8f;

#define D_DIM 2048
#define F_DIM 512
#define B_DIM 32

// ---------------------------------------------------------------------------
// Kernel 1: abs_sum[j] = sum_k |s[j] - s[k]|
// ---------------------------------------------------------------------------
__global__ void k_abs_sum(const float* __restrict__ s, float* __restrict__ abs_sum) {
    __shared__ float sh[D_DIM];
    const int tid = threadIdx.x;
    for (int i = tid; i < D_DIM; i += 256) sh[i] = s[i];
    __syncthreads();
    const int j = blockIdx.x * 256 + tid;
    const float sj = sh[j];
    float acc = 0.0f;
    #pragma unroll 8
    for (int k = 0; k < D_DIM; ++k) acc += fabsf(sj - sh[k]);
    abs_sum[j] = acc;
}

// ---------------------------------------------------------------------------
// Kernel 2: per softmax-row d: rowmax[d], rowsum[d] (sum of exp(l - max))
// logits[d, j] = (2047 - 2d) * s[j] - abs_sum[j]   (TAU == 1)
// ---------------------------------------------------------------------------
__global__ void k_row_stats(const float* __restrict__ s,
                            const float* __restrict__ abs_sum,
                            float* __restrict__ rowmax,
                            float* __restrict__ rowsum) {
    __shared__ float red[256];
    const int d   = blockIdx.x;
    const int tid = threadIdx.x;
    const float scal = 2047.0f - 2.0f * (float)d;

    float lmax = -3.4e38f;
    for (int j = tid; j < D_DIM; j += 256) {
        float l = scal * s[j] - abs_sum[j];
        lmax = fmaxf(lmax, l);
    }
    red[tid] = lmax;
    __syncthreads();
    for (int off = 128; off > 0; off >>= 1) {
        if (tid < off) red[tid] = fmaxf(red[tid], red[tid + off]);
        __syncthreads();
    }
    const float m = red[0];
    __syncthreads();

    float lsum = 0.0f;
    for (int j = tid; j < D_DIM; j += 256) {
        float l = scal * s[j] - abs_sum[j];
        lsum += __expf(l - m);
    }
    red[tid] = lsum;
    __syncthreads();
    for (int off = 128; off > 0; off >>= 1) {
        if (tid < off) red[tid] += red[tid + off];
        __syncthreads();
    }
    if (tid == 0) { rowmax[d] = m; rowsum[d] = red[0]; }
}

// ---------------------------------------------------------------------------
// Kernel 3: materialize PT = P^T in f16 (PT[j*D + d] = P[d][j]),
// 64x64 tiles, writes coalesced along d.
// ---------------------------------------------------------------------------
__global__ void k_write_pt(const float* __restrict__ s,
                           const float* __restrict__ abs_sum,
                           const float* __restrict__ rowmax,
                           const float* __restrict__ rowsum,
                           _Float16* __restrict__ PT) {
    __shared__ float sj_sh[64];
    __shared__ float as_sh[64];
    const int Dbase = blockIdx.x * 64;
    const int Jbase = blockIdx.y * 64;
    const int tid = threadIdx.x;
    if (tid < 64) {
        sj_sh[tid] = s[Jbase + tid];
        as_sh[tid] = abs_sum[Jbase + tid];
    }
    __syncthreads();
    const int dl = tid & 63;
    const int jl = tid >> 6;          // 0..3
    const int d  = Dbase + dl;
    const float scal = 2047.0f - 2.0f * (float)d;
    const float m    = rowmax[d];
    const float rinv = 1.0f / rowsum[d];
    #pragma unroll
    for (int jj = 0; jj < 16; ++jj) {
        const int jo = jl * 16 + jj;
        const int j  = Jbase + jo;
        const float v = __expf(scal * sj_sh[jo] - as_sh[jo] - m) * rinv;
        PT[(size_t)j * D_DIM + d] = (_Float16)v;   // 64 consecutive halves, coalesced
    }
}

// ---------------------------------------------------------------------------
// Kernel 4: out[b] = PT(MxK = j x d, f16, row-major)  @  x[b](K x N = d x f)
// Block tile 128(M) x 128(N), K-step 32; 8 waves as 2(M) x 4(N),
// each wave: 4 M-subtiles x 2 N-subtiles of v_wmma_f32_16x16x32_f16.
// X chunk is transposed in registers at staging time, so BOTH operand
// fragments are two ds_load_b128 each. Software-pipelined K loop.
// ---------------------------------------------------------------------------
#define AS_STRIDE 40   // halves; 80B rows -> each 16-lane half hits all 64 banks
#define XT_STRIDE 40   // halves; Xt[n][k] layout, same property

__global__ __launch_bounds__(256)
void k_gemm(const _Float16* __restrict__ PT,
            const float* __restrict__ x,
            float* __restrict__ out) {
    __shared__ _Float16 As[128 * AS_STRIDE];  // [row j 0..127][k 0..31]
    __shared__ _Float16 Xt[128 * XT_STRIDE];  // [col n 0..127][k 0..31]  (transposed chunk)

    const int tid  = threadIdx.x;
    const int lane = tid & 31;
    const int wave = tid >> 5;
    const int wm   = (wave & 1) * 64;   // wave M offset
    const int wn   = (wave >> 1) * 32;  // wave N offset
    const int lr   = lane & 15;         // row/col within 16x16 tile
    const int lh   = lane >> 4;         // which 16-lane half

    const int Mbase = blockIdx.x * 128;
    const int Nbase = blockIdx.y * 128;
    const int b     = blockIdx.z;

    const size_t xbase = (size_t)b * D_DIM * F_DIM + (size_t)Nbase;

    // A staging: 2 threads per row, 16 halves each
    const int ar = tid >> 1;
    const int ah = tid & 1;
    const _Float16* pta = PT + (size_t)(Mbase + ar) * D_DIM + ah * 16;

    // X staging: thread owns column n, k-halfrange kh*16..kh*16+15
    const int xn = tid & 127;
    const int kh = tid >> 7;
    const float* xp = x + xbase + (size_t)(kh * 16) * F_DIM + xn;

    v8f acc[4][2] = {};

    // ---- software pipeline: preload chunk k0 = 0 into registers ----
    uint4 areg0, areg1;
    float xreg[16];
    {
        const uint4* src = (const uint4*)(pta);
        areg0 = src[0];
        areg1 = src[1];
        #pragma unroll
        for (int i = 0; i < 16; ++i) xreg[i] = xp[(size_t)i * F_DIM];
    }

    for (int k0 = 0; k0 < D_DIM; k0 += 32) {
        // ---- store staged registers to LDS ----
        {
            uint4* dst = (uint4*)(As + ar * AS_STRIDE + ah * 16);
            dst[0] = areg0;
            dst[1] = areg1;
            v8h p0, p1;
            #pragma unroll
            for (int i = 0; i < 8; ++i) { p0[i] = (_Float16)xreg[i]; p1[i] = (_Float16)xreg[8 + i]; }
            *(v8h*)(Xt + xn * XT_STRIDE + kh * 16)     = p0;
            *(v8h*)(Xt + xn * XT_STRIDE + kh * 16 + 8) = p1;
        }
        __syncthreads();

        // ---- issue global loads for next chunk (overlaps with WMMA below) ----
        if (k0 + 32 < D_DIM) {
            const uint4* src = (const uint4*)(pta + (k0 + 32));
            areg0 = src[0];
            areg1 = src[1];
            const float* xq = xp + (size_t)(k0 + 32) * F_DIM;
            #pragma unroll
            for (int i = 0; i < 16; ++i) xreg[i] = xq[(size_t)i * F_DIM];
        }

        // ---- A fragments: lane = row M, K = {koff..koff+7, 16+koff..16+koff+7} ----
        v16h afrag[4];
        #pragma unroll
        for (int t = 0; t < 4; ++t) {
            const _Float16* base = As + (wm + t * 16 + lr) * AS_STRIDE + lh * 8;
            v8h lo = *(const v8h*)(base);        // K = koff..koff+7
            v8h hi = *(const v8h*)(base + 16);   // K = 16+koff..16+koff+7
            afrag[t] = __builtin_shufflevector(lo, hi,
                0, 1, 2, 3, 4, 5, 6, 7, 8, 9, 10, 11, 12, 13, 14, 15);
        }
        // ---- B fragments: lane = col N, K = {koff..koff+7, 16+koff..16+koff+7} ----
        v16h bfrag[2];
        #pragma unroll
        for (int u = 0; u < 2; ++u) {
            const _Float16* base = Xt + (wn + u * 16 + lr) * XT_STRIDE + lh * 8;
            v8h lo = *(const v8h*)(base);        // K = koff..koff+7
            v8h hi = *(const v8h*)(base + 16);   // K = 16+koff..16+koff+7
            bfrag[u] = __builtin_shufflevector(lo, hi,
                0, 1, 2, 3, 4, 5, 6, 7, 8, 9, 10, 11, 12, 13, 14, 15);
        }
        // ---- 8 WMMAs per wave per K-step ----
        #pragma unroll
        for (int t = 0; t < 4; ++t)
            #pragma unroll
            for (int u = 0; u < 2; ++u)
                acc[t][u] = __builtin_amdgcn_wmma_f32_16x16x32_f16(
                    false, afrag[t], false, bfrag[u],
                    (short)0, acc[t][u], false, false);

        __syncthreads();   // all reads done before next store phase
    }

    // ---- epilogue: C/D layout -> lane (lh,lr): rows lh*8+v, col lr ----
    #pragma unroll
    for (int t = 0; t < 4; ++t)
        #pragma unroll
        for (int u = 0; u < 2; ++u) {
            const int r0 = Mbase + wm + t * 16 + lh * 8;
            const int c0 = Nbase + wn + u * 16 + lr;
            float* o = out + ((size_t)b * D_DIM + r0) * F_DIM + c0;
            #pragma unroll
            for (int v = 0; v < 8; ++v) o[(size_t)v * F_DIM] = acc[t][u][v];
        }
}

// ---------------------------------------------------------------------------
// Workspace layout (needs 32 KB + 8 MB):
//   [0, 8KB)      abs_sum  f32[2048]
//   [8KB, 16KB)   rowmax   f32[2048]
//   [16KB, 24KB)  rowsum   f32[2048]
//   [32KB, +8MB)  PT       f16[2048*2048]
// ---------------------------------------------------------------------------
extern "C" void kernel_launch(void* const* d_in, const int* in_sizes, int n_in,
                              void* d_out, int out_size, void* d_ws, size_t ws_size,
                              hipStream_t stream) {
    const float* x = (const float*)d_in[0];            // [32, 2048, 512] f32
    const float* s = (const float*)d_in[1];            // [2048] f32
    float* out = (float*)d_out;                        // [32, 2048, 512] f32

    char* ws = (char*)d_ws;
    float* abs_sum = (float*)ws;
    float* rowmax  = abs_sum + D_DIM;
    float* rowsum  = rowmax + D_DIM;
    _Float16* PT   = (_Float16*)(ws + 32768);

    k_abs_sum <<<D_DIM / 256, 256, 0, stream>>>(s, abs_sum);
    k_row_stats<<<D_DIM,       256, 0, stream>>>(s, abs_sum, rowmax, rowsum);
    k_write_pt <<<dim3(D_DIM / 64, D_DIM / 64), 256, 0, stream>>>(s, abs_sum, rowmax, rowsum, PT);
    k_gemm     <<<dim3(D_DIM / 128, F_DIM / 128, B_DIM), 256, 0, stream>>>(PT, x, out);
}
